// InterMultiHeadAttention_66271345377349
// MI455X (gfx1250) — compile-verified, matrix operand order checked by reading, not run on hardware
//
#include <hip/hip_runtime.h>
#include <hip/hip_bf16.h>

typedef __attribute__((ext_vector_type(16))) __bf16 v16bf;
typedef __attribute__((ext_vector_type(4)))  __bf16 v4bf;
typedef __attribute__((ext_vector_type(8)))  float  v8f;

#define WMMA_BF16(a, b, c) \
    __builtin_amdgcn_wmma_f32_16x16x32_bf16(false, (a), false, (b), (short)0, (c), false, false)

// ---------------------------------------------------------------------------
// Fragment loader for V_WMMA_F32_16X16X32_BF16 (wave32).
// 16-bit A layout (16x32, M x K):
//   lanes 0-15 : row M = lane,    elems 0..7 = K 0..7,  elems 8..15 = K 16..23
//   lanes 16-31: row M = lane-16, elems 0..7 = K 8..15, elems 8..15 = K 24..31
// B (32x16, K x N) is symmetric with "row" := column N (contiguous in k when
// the source stores N-rows with leading dim = k-stride).
// ---------------------------------------------------------------------------
__device__ __forceinline__ v16bf frag_from_bf16(const __bf16* __restrict__ src,
                                                int ld, int row0, int k0) {
    const int lane = threadIdx.x & 31;
    const int r    = lane & 15;
    const int g    = lane >> 4;
    const __bf16* p = src + (size_t)(row0 + r) * ld + k0 + g * 8;
    v16bf a;
#pragma unroll
    for (int i = 0; i < 8; ++i) a[i]     = p[i];
#pragma unroll
    for (int i = 0; i < 8; ++i) a[8 + i] = p[16 + i];
    return a;
}

// ---------------------------------------------------------------------------
// One-pass f32 -> bf16 conversion (vector of 4 per thread).
// ---------------------------------------------------------------------------
__global__ __launch_bounds__(256) void cvt_bf16_kernel(const float* __restrict__ in,
                                                       __bf16* __restrict__ out, int n4) {
    const int i = blockIdx.x * blockDim.x + threadIdx.x;
    if (i < n4) {
        const float4 f = ((const float4*)in)[i];
        v4bf o;
        o[0] = (__bf16)f.x; o[1] = (__bf16)f.y; o[2] = (__bf16)f.z; o[3] = (__bf16)f.w;
        ((v4bf*)out)[i] = o;
    }
}

// mask (int 0/1) -> additive f32 bias {-1e9, 0}: folds masking into one v_fmac.
__global__ __launch_bounds__(256) void mask_bias_kernel(const int* __restrict__ mask,
                                                        float* __restrict__ mb, int n) {
    const int i = blockIdx.x * blockDim.x + threadIdx.x;
    if (i < n) mb[i] = (mask[i] == 0) ? -1e9f : 0.0f;
}

// ---------------------------------------------------------------------------
// Projection GEMM: Y = X @ W^T + bias.  X:[M,512] bf16, W:[512,512] bf16.
// Each wave computes a 16x64 strip; register double-buffered k-loop so the
// next step's fragment loads are in flight while the current WMMAs execute.
// mode 0: -> q  bf16 [B,H,Lq,dk]
// mode 1: -> k  bf16 [H,KV,dk]
// mode 2: -> vT bf16 [H,dk,KV]   (transposed store)
// mode 3: -> f32 out [M,512]
// ---------------------------------------------------------------------------
__global__ __launch_bounds__(128) void proj_gemm_kernel(
    const __bf16* __restrict__ X, const __bf16* __restrict__ W,
    const float* __restrict__ bias, void* __restrict__ out, int mode) {
    const int wave  = threadIdx.x >> 5;
    const int m0    = blockIdx.x * 16;
    const int nbase = blockIdx.y * 256 + wave * 64;

    v8f acc[4] = {v8f{}, v8f{}, v8f{}, v8f{}};

    // prologue: fragments for k0 = 0
    v16bf a = frag_from_bf16(X, 512, m0, 0);
    v16bf b[4];
#pragma unroll
    for (int j = 0; j < 4; ++j) b[j] = frag_from_bf16(W, 512, nbase + j * 16, 0);

#pragma unroll
    for (int k0 = 32; k0 <= 512; k0 += 32) {
        v16bf an = a;
        v16bf bn[4] = {b[0], b[1], b[2], b[3]};
        if (k0 < 512) {                 // issue next-step loads first
            an = frag_from_bf16(X, 512, m0, k0);
#pragma unroll
            for (int j = 0; j < 4; ++j) bn[j] = frag_from_bf16(W, 512, nbase + j * 16, k0);
        }
#pragma unroll
        for (int j = 0; j < 4; ++j) acc[j] = WMMA_BF16(a, b[j], acc[j]);
        a = an;
#pragma unroll
        for (int j = 0; j < 4; ++j) b[j] = bn[j];
    }

    const int lane = threadIdx.x & 31;
    const int nc   = lane & 15;
    const int roff = (lane >> 4) * 8;

#pragma unroll
    for (int j = 0; j < 4; ++j) {
        const int n  = nbase + j * 16 + nc;
        const float bv = bias[n];
        const int h = n >> 6, d = n & 63;
#pragma unroll
        for (int r = 0; r < 8; ++r) {
            const int m = m0 + r + roff;
            const float val = acc[j][r] + bv;
            if (mode == 0) {
                const int b_ = m >> 8, lq = m & 255;
                ((__bf16*)out)[(((size_t)(b_ * 8 + h) * 256 + lq) * 64) + d] = (__bf16)val;
            } else if (mode == 1) {
                ((__bf16*)out)[((size_t)h * 4096 + m) * 64 + d] = (__bf16)val;
            } else if (mode == 2) {
                ((__bf16*)out)[((size_t)(h * 64 + d)) * 4096 + m] = (__bf16)val;
            } else {
                ((float*)out)[(size_t)m * 512 + n] = val;
            }
        }
    }
}

// ---------------------------------------------------------------------------
// Flash-style attention. One wave per (b,h, 16-query tile).
// S^T = K_chunk(16x64) @ Q^T(64x16): the f32 C-layout of S^T maps exactly onto
// the bf16 B-fragment layout of the next WMMA (O^T = V^T @ P^T), so the
// probabilities never cross lanes. Softmax state is per-lane (one query row).
// K fragments are double-buffered; V fragments load before the softmax VALU
// block so exp/shuffle work hides their latency.
// ---------------------------------------------------------------------------
__global__ __launch_bounds__(128) void attn_kernel(
    const __bf16* __restrict__ q, const __bf16* __restrict__ k,
    const __bf16* __restrict__ vT, const float* __restrict__ sim,
    const float* __restrict__ mbias, __bf16* __restrict__ x) {
    const int wave = threadIdx.x >> 5;
    const int lqt  = blockIdx.x * 4 + wave;       // 0..15
    const int bh   = blockIdx.y;                  // 0..63
    const int b    = bh >> 3, h = bh & 7;
    const int lq0  = lqt * 16;

    const int lane = threadIdx.x & 31;
    const int col  = lane & 15;                   // query row owned by this lane
    const int g    = lane >> 4;

    const __bf16* qh = q  + ((size_t)(b * 8 + h) * 256 + lq0) * 64;
    const __bf16* kh = k  + (size_t)h * 4096 * 64;
    const __bf16* vh = vT + (size_t)h * 64 * 4096;
    const float*  sm = sim + ((size_t)b * 256 + lq0 + col) * 64;

    // Q^T as B fragments (loaded once): B[k=d][n=m] = q[m][d]
    const v16bf bq0 = frag_from_bf16(qh, 64, 0, 0);
    const v16bf bq1 = frag_from_bf16(qh, 64, 0, 32);

    v8f o[4] = {v8f{}, v8f{}, v8f{}, v8f{}};
    float runmax = -1e30f;
    float runsum = 0.0f;

    // prologue: K fragments for kv0 = 0
    v16bf ka[4] = {frag_from_bf16(kh, 64, 0, 0),  frag_from_bf16(kh, 64, 0, 32),
                   frag_from_bf16(kh, 64, 16, 0), frag_from_bf16(kh, 64, 16, 32)};

    for (int kv0 = 0; kv0 < 4096; kv0 += 32) {
        // --- S^T tiles from resident K fragments ---
        v8f s0 = {}, s1 = {};
        s0 = WMMA_BF16(ka[0], bq0, s0);
        s0 = WMMA_BF16(ka[1], bq1, s0);
        s1 = WMMA_BF16(ka[2], bq0, s1);
        s1 = WMMA_BF16(ka[3], bq1, s1);

        // --- issue next K-chunk loads (double buffer) ---
        if (kv0 + 32 < 4096) {
            ka[0] = frag_from_bf16(kh, 64, kv0 + 32, 0);
            ka[1] = frag_from_bf16(kh, 64, kv0 + 32, 32);
            ka[2] = frag_from_bf16(kh, 64, kv0 + 48, 0);
            ka[3] = frag_from_bf16(kh, 64, kv0 + 48, 32);
        }

        // --- issue V^T fragment loads; softmax VALU below hides latency ---
        v16bf av[4];
#pragma unroll
        for (int t = 0; t < 4; ++t) av[t] = frag_from_bf16(vh, 4096, t * 16, kv0);

        // --- sv = s * (scale*sim) + mbias  (mbias in {0,-1e9}) ---
        const float f = 0.125f * sm[kv0 >> 6];
        float sv[16];
#pragma unroll
        for (int r = 0; r < 8; ++r) {
            const int kva = kv0 + r + g * 8;
            sv[r]     = fmaf(s0[r], f, mbias[kva]);
            sv[8 + r] = fmaf(s1[r], f, mbias[kva + 16]);
        }

        // --- online softmax (state per lane == per query row) ---
        float cmax = sv[0];
#pragma unroll
        for (int i = 1; i < 16; ++i) cmax = fmaxf(cmax, sv[i]);
        cmax = fmaxf(cmax, __shfl_xor(cmax, 16, 32));
        const float nm   = fmaxf(runmax, cmax);
        const float corr = __expf(runmax - nm);
        runmax = nm;

        float csum = 0.0f;
#pragma unroll
        for (int i = 0; i < 16; ++i) { sv[i] = __expf(sv[i] - nm); csum += sv[i]; }
        csum += __shfl_xor(csum, 16, 32);
        runsum = runsum * corr + csum;

#pragma unroll
        for (int t = 0; t < 4; ++t)
#pragma unroll
            for (int r = 0; r < 8; ++r) o[t][r] *= corr;

        // --- P^T as B fragment: S^T C-layout maps directly onto B layout ---
        v16bf bp;
#pragma unroll
        for (int i = 0; i < 8; ++i) bp[i]     = (__bf16)sv[i];
#pragma unroll
        for (int i = 0; i < 8; ++i) bp[8 + i] = (__bf16)sv[8 + i];

        // --- O^T += V^T_chunk @ P^T ---
#pragma unroll
        for (int t = 0; t < 4; ++t) o[t] = WMMA_BF16(av[t], bp, o[t]);
    }

    // --- normalize and store x[b][lq][h*64+d] as bf16 for the O-projection ---
    const float inv = 1.0f / runsum;
    __bf16* xr = x + ((size_t)b * 256 + lq0 + col) * 512 + h * 64;
#pragma unroll
    for (int t = 0; t < 4; ++t)
#pragma unroll
        for (int r = 0; r < 8; ++r)
            xr[t * 16 + r + g * 8] = (__bf16)(o[t][r] * inv);
}

// ---------------------------------------------------------------------------
extern "C" void kernel_launch(void* const* d_in, const int* in_sizes, int n_in,
                              void* d_out, int out_size, void* d_ws, size_t ws_size,
                              hipStream_t stream) {
    const float* query = (const float*)d_in[0];
    const float* key   = (const float*)d_in[1];
    const float* value = (const float*)d_in[2];
    const int*   mask  = (const int*)d_in[3];
    const float* sim   = (const float*)d_in[4];
    const float* Wq = (const float*)d_in[5];  const float* bq = (const float*)d_in[6];
    const float* Wk = (const float*)d_in[7];  const float* bk = (const float*)d_in[8];
    const float* Wv = (const float*)d_in[9];  const float* bv = (const float*)d_in[10];
    const float* Wo = (const float*)d_in[11]; const float* bo = (const float*)d_in[12];

    // Workspace layout (~24 MB total)
    __bf16* qb   = (__bf16*)d_ws;                    // 1,048,576  (2 MB)
    __bf16* kb   = qb   + (size_t)1048576;           // 2,097,152  (4 MB)
    __bf16* vb   = kb   + (size_t)2097152;           // 2,097,152  (4 MB)
    __bf16* xb   = vb   + (size_t)2097152;           // 1,048,576  (2 MB)
    __bf16* wq16 = xb   + (size_t)1048576;           // 262,144    (0.5 MB)
    __bf16* wk16 = wq16 + (size_t)262144;
    __bf16* wv16 = wk16 + (size_t)262144;
    __bf16* wo16 = wv16 + (size_t)262144;
    __bf16* xq16 = wo16 + (size_t)262144;            // 1,048,576  (2 MB)
    __bf16* xk16 = xq16 + (size_t)1048576;           // 2,097,152  (4 MB)
    __bf16* xv16 = xk16 + (size_t)2097152;           // 2,097,152  (4 MB)
    float*  mb   = (float*)(xv16 + (size_t)2097152); // 4096 f32   (16 KB)

    // Pre-convert weights + activations to bf16; build mask bias
    const int W4 = 512 * 512 / 4, Q4 = 2048 * 512 / 4, K4 = 4096 * 512 / 4;
    cvt_bf16_kernel<<<(W4 + 255) / 256, 256, 0, stream>>>(Wq, wq16, W4);
    cvt_bf16_kernel<<<(W4 + 255) / 256, 256, 0, stream>>>(Wk, wk16, W4);
    cvt_bf16_kernel<<<(W4 + 255) / 256, 256, 0, stream>>>(Wv, wv16, W4);
    cvt_bf16_kernel<<<(W4 + 255) / 256, 256, 0, stream>>>(Wo, wo16, W4);
    cvt_bf16_kernel<<<(Q4 + 255) / 256, 256, 0, stream>>>(query, xq16, Q4);
    cvt_bf16_kernel<<<(K4 + 255) / 256, 256, 0, stream>>>(key,   xk16, K4);
    cvt_bf16_kernel<<<(K4 + 255) / 256, 256, 0, stream>>>(value, xv16, K4);
    mask_bias_kernel<<<16, 256, 0, stream>>>(mask, mb, 4096);

    // QKV projections (grid.y = 512/256 = 2; each wave does 16x64)
    proj_gemm_kernel<<<dim3(128, 2), 128, 0, stream>>>(xq16, wq16, bq, (void*)qb, 0);
    proj_gemm_kernel<<<dim3(256, 2), 128, 0, stream>>>(xk16, wk16, bk, (void*)kb, 1);
    proj_gemm_kernel<<<dim3(256, 2), 128, 0, stream>>>(xv16, wv16, bv, (void*)vb, 2);

    // Attention: 16 lq-tiles x 64 (b,h) pairs; 4 waves/block
    attn_kernel<<<dim3(4, 64), 128, 0, stream>>>(qb, kb, vb, sim, mb, xb);

    // Output projection -> f32 d_out
    proj_gemm_kernel<<<dim3(128, 2), 128, 0, stream>>>(xb, wo16, bo, d_out, 3);
}
